// TransformerBlock_5866925326359
// MI455X (gfx1250) — matrix operand order repaired and checked by reading, not compiled, not run on hardware
//
#include <hip/hip_runtime.h>
#include <hip/hip_bf16.h>
#include <math.h>

typedef __bf16 bf16;
typedef float v8f   __attribute__((ext_vector_type(8)));
typedef bf16  v16bf __attribute__((ext_vector_type(16)));
typedef unsigned int u32x4 __attribute__((ext_vector_type(4)));
typedef bf16  bf16x4 __attribute__((ext_vector_type(4)));

union Frag  { v16bf v; u32x4 q[2]; };
union Pack8 { u32x4 q; bf16 e[8]; };

// ---------------------------------------------------------------------------
// CDNA5 async global->LDS copies (ASYNCcnt-tracked), with sync fallback
// ---------------------------------------------------------------------------
#if __has_builtin(__builtin_amdgcn_global_load_async_to_lds_b128)
#define HAS_ASYNC 1
typedef int v4i __attribute__((ext_vector_type(4)));
typedef __attribute__((address_space(1))) v4i as1_v4i;
typedef __attribute__((address_space(3))) v4i as3_v4i;
__device__ __forceinline__ void async_cp16(const void* g, void* l) {
    // flat->AS1: identical numeric value for global memory.
    // flat->AS3: CDNA5 maps LDS flat addresses via addr[31:0] (ISA 10.2), so
    // truncation == addrspacecast.
    __builtin_amdgcn_global_load_async_to_lds_b128(
        (as1_v4i*)(unsigned long long)g,
        (as3_v4i*)(unsigned)(unsigned long long)l, 0, 0);
}
#if __has_builtin(__builtin_amdgcn_s_wait_asynccnt)
#define WAIT_ASYNC(n) __builtin_amdgcn_s_wait_asynccnt(n)
#else
#define WAIT_ASYNC(n) asm volatile("s_wait_asynccnt %0" ::"i"(n) : "memory")
#endif
#else
#define HAS_ASYNC 0
#define WAIT_ASYNC(n) ((void)0)
#endif

// ---------------------------------------------------------------------------
// fp32 [K][N] -> bf16 transposed [N][K]  (weights; tiled via LDS)
// ---------------------------------------------------------------------------
__global__ __launch_bounds__(256)
void cvt_transpose_bf16_kernel(const float* __restrict__ src, bf16* __restrict__ dst,
                               int K, int N)
{
    __shared__ float t[32][33];
    const int k0 = blockIdx.y * 32;
    const int n0 = blockIdx.x * 32;
    const int tx = threadIdx.x & 31;
    const int ty = threadIdx.x >> 5;
    for (int i = ty; i < 32; i += 8)
        t[i][tx] = src[(size_t)(k0 + i) * N + n0 + tx];
    __syncthreads();
    for (int i = ty; i < 32; i += 8)
        dst[(size_t)(n0 + i) * K + k0 + tx] = (bf16)t[tx][i];
}

// ---------------------------------------------------------------------------
// LayerNorm: one block (256 threads = 8 waves) per row of 1024, bf16 output
// ---------------------------------------------------------------------------
__global__ __launch_bounds__(256)
void layernorm_bf16_kernel(const float* __restrict__ x, const float* __restrict__ g,
                           const float* __restrict__ beta, bf16* __restrict__ out,
                           int Ddim)
{
    __shared__ float s_sum[8], s_sq[8];
    const int row = blockIdx.x;
    const float* xr = x + (size_t)row * Ddim;
    const int tid = threadIdx.x;

    float vals[4];
    float lsum = 0.f, lsq = 0.f;
    for (int i = 0; i < 4; ++i) {
        vals[i] = xr[tid + i * 256];
        lsum += vals[i];
        lsq  += vals[i] * vals[i];
    }
    for (int m = 16; m >= 1; m >>= 1) {
        lsum += __shfl_xor(lsum, m, 32);
        lsq  += __shfl_xor(lsq,  m, 32);
    }
    if ((tid & 31) == 0) { s_sum[tid >> 5] = lsum; s_sq[tid >> 5] = lsq; }
    __syncthreads();
    float tsum = 0.f, tsq = 0.f;
    for (int i = 0; i < 8; ++i) { tsum += s_sum[i]; tsq += s_sq[i]; }
    const float mean = tsum / (float)Ddim;
    const float var  = tsq / (float)Ddim - mean * mean;
    const float rstd = rsqrtf(var + 1e-5f);
    for (int i = 0; i < 4; ++i) {
        int c = tid + i * 256;
        out[(size_t)row * Ddim + c] = (bf16)((vals[i] - mean) * rstd * g[c] + beta[c]);
    }
}

// ---------------------------------------------------------------------------
// bf16 WMMA GEMM: C[M,N] = A[M,K] @ Bt[N,K]^T (+bias)(GELU?)(+resid)
// 128x128 block tile, K-step 32, double-buffered async LDS staging.
// 8 waves: 4 (M) x 2 (N); wave tile 32x64 -> 8 WMMAs per K-step.
// ---------------------------------------------------------------------------
#define BM 128
#define BN 128
#define BK 32
#define LDT 40

template<bool BIAS, bool GELU, bool RESID, bool OUTF, bool OUTB>
__global__ __launch_bounds__(256)
void gemm_bf16_kernel(const bf16* __restrict__ A, const bf16* __restrict__ Bt,
                      const float* __restrict__ bias, const float* __restrict__ resid,
                      float* __restrict__ outF, bf16* __restrict__ outB,
                      int Mdim, int Ndim, int Kdim)
{
    __shared__ __align__(16) bf16 As[2][BM][LDT];
    __shared__ __align__(16) bf16 Bs[2][BN][LDT];   // Bs[n][k]

    const int tid      = threadIdx.x;
    const int lane     = tid & 31;
    const int wave     = tid >> 5;
    const int wm       = wave & 3;
    const int wn       = wave >> 2;
    const int lane16   = lane & 15;
    const int laneHalf = lane >> 4;

    const int bm0 = blockIdx.y * BM;
    const int bn0 = blockIdx.x * BN;

    v8f acc[2][4];
    for (int i = 0; i < 2; ++i)
        for (int j = 0; j < 4; ++j) acc[i][j] = {};

    // staging: 2 threads per row, 16 bf16 each, for both A and Bt tiles
    const int srow = tid >> 1;
    const int scol = (tid & 1) * 16;
    const int gra  = bm0 + srow;
    const size_t aRow = (size_t)((gra < Mdim) ? gra : 0);  // safe dummy row, EXEC stays full
    const size_t bRow = (size_t)(bn0 + srow);

    auto stage = [&](int buf, int k0) {
#if HAS_ASYNC
        async_cp16(A  + aRow * Kdim + k0 + scol,     &As[buf][srow][scol]);
        async_cp16(A  + aRow * Kdim + k0 + scol + 8, &As[buf][srow][scol + 8]);
        async_cp16(Bt + bRow * Kdim + k0 + scol,     &Bs[buf][srow][scol]);
        async_cp16(Bt + bRow * Kdim + k0 + scol + 8, &Bs[buf][srow][scol + 8]);
#else
        const u32x4* sa = (const u32x4*)(A  + aRow * Kdim + k0 + scol);
        const u32x4* sb = (const u32x4*)(Bt + bRow * Kdim + k0 + scol);
        u32x4 a0 = sa[0], a1 = sa[1], b0 = sb[0], b1 = sb[1];
        *(u32x4*)&As[buf][srow][scol]     = a0;
        *(u32x4*)&As[buf][srow][scol + 8] = a1;
        *(u32x4*)&Bs[buf][srow][scol]     = b0;
        *(u32x4*)&Bs[buf][srow][scol + 8] = b1;
#endif
    };

    const int nt = Kdim / BK;
    stage(0, 0);
    for (int t = 0; t < nt; ++t) {
        const int cur = t & 1;
        if (t + 1 < nt) {
            stage(cur ^ 1, (t + 1) * BK);  // prefetch next tile
            WAIT_ASYNC(4);                 // drain current tile, keep prefetch in flight
        } else {
            WAIT_ASYNC(0);
        }
        __syncthreads();

        Frag a[2];
        for (int mi = 0; mi < 2; ++mi) {
            const int r   = wm * 32 + mi * 16 + lane16;
            const int off = laneHalf * 8;
            a[mi].q[0] = *(const u32x4*)&As[cur][r][off];
            a[mi].q[1] = *(const u32x4*)&As[cur][r][off + 16];
        }
        Frag b[4];
        for (int ni = 0; ni < 4; ++ni) {
            const int r   = wn * 64 + ni * 16 + lane16;
            const int off = laneHalf * 8;
            b[ni].q[0] = *(const u32x4*)&Bs[cur][r][off];
            b[ni].q[1] = *(const u32x4*)&Bs[cur][r][off + 16];
        }
        for (int mi = 0; mi < 2; ++mi)
            for (int ni = 0; ni < 4; ++ni)
                acc[mi][ni] = __builtin_amdgcn_wmma_f32_16x16x32_bf16(
                    false, a[mi].v, false, b[ni].v, (short)0, acc[mi][ni], false, false);
        __syncthreads();
    }

    // ---- branch-free templated epilogue ----
    for (int mi = 0; mi < 2; ++mi) {
        for (int ni = 0; ni < 4; ++ni) {
            const int gcol = bn0 + wn * 64 + ni * 16 + lane16;
            float bval = 0.0f;
            if constexpr (BIAS) bval = bias[gcol];
            for (int r = 0; r < 8; ++r) {
                const int grow = bm0 + wm * 32 + mi * 16 + laneHalf * 8 + r;
                if (grow >= Mdim) continue;
                float v = acc[mi][ni][r] + bval;
                if constexpr (GELU)  v = 0.5f * v * (1.0f + erff(v * 0.70710678118654752f));
                if constexpr (RESID) v += resid[(size_t)grow * Ndim + gcol];
                if constexpr (OUTF)  outF[(size_t)grow * Ndim + gcol] = v;
                if constexpr (OUTB)  outB[(size_t)grow * Ndim + gcol] = (bf16)v;
            }
        }
    }
}

// ---------------------------------------------------------------------------
// Flash-style causal attention, dk = 64, q-tile 128, kv-tile 128.
// 8 waves; each wave owns 16 q rows. Q kept as register A-fragments.
// ---------------------------------------------------------------------------
__global__ __launch_bounds__(256)
void attention_kernel(const bf16* __restrict__ Q, const bf16* __restrict__ Km,
                      const bf16* __restrict__ Vm, bf16* __restrict__ O,
                      int Sdim, int Dmodel)
{
    __shared__ __align__(16) bf16 Qs[128][72];
    __shared__ __align__(16) bf16 Ks[128][72];      // [kv][dk] (B-layout for QK^T)
    __shared__ __align__(16) bf16 Vt[64][136];      // [dk][kv] (transposed V)
    __shared__ __align__(16) bf16 Ps[8][16][136];   // per-wave softmax tile

    const int tid      = threadIdx.x;
    const int lane     = tid & 31;
    const int wave     = tid >> 5;
    const int lane16   = lane & 15;
    const int laneHalf = lane >> 4;

    const int q0 = blockIdx.x * 128;
    const int h  = blockIdx.y;
    const int b  = blockIdx.z;

    const size_t headOff   = (size_t)h * 64;
    const size_t rowStride = (size_t)Dmodel;
    const size_t baseBS    = (size_t)b * Sdim;

    // ---- stage Q tile (async; OOB rows read safe dummy row, masked later) ----
    {
        const int r  = tid >> 1;
        const int c0 = (tid & 1) * 32;
        const int s  = q0 + r;
        const size_t sr = (size_t)(s < Sdim ? s : 0);
#if HAS_ASYNC
        for (int i = 0; i < 4; ++i)
            async_cp16(Q + (baseBS + sr) * rowStride + headOff + c0 + i * 8,
                       &Qs[r][c0 + i * 8]);
#else
        const u32x4* src = (const u32x4*)(Q + (baseBS + sr) * rowStride + headOff + c0);
        for (int i = 0; i < 4; ++i) *(u32x4*)&Qs[r][c0 + i * 8] = src[i];
#endif
    }
    WAIT_ASYNC(0);
    __syncthreads();

    Frag qf[2];
    {
        const int r  = wave * 16 + lane16;
        const int ho = laneHalf * 8;
        for (int kk = 0; kk < 2; ++kk) {
            qf[kk].q[0] = *(const u32x4*)&Qs[r][kk * 32 + ho];
            qf[kk].q[1] = *(const u32x4*)&Qs[r][kk * 32 + 16 + ho];
        }
    }

    v8f oacc[4];
    for (int i = 0; i < 4; ++i) oacc[i] = {};
    float mrow[8], lrow[8];
    for (int r = 0; r < 8; ++r) { mrow[r] = -INFINITY; lrow[r] = 0.0f; }

    int kvEnd = q0 + 128; if (kvEnd > Sdim) kvEnd = Sdim;

    for (int kv0 = 0; kv0 < kvEnd; kv0 += 128) {
        // ---- stage K tile (async) ----
        {
            const int r  = tid >> 1;
            const int c0 = (tid & 1) * 32;
            const int s  = kv0 + r;
            const size_t sr = (size_t)(s < Sdim ? s : 0);
#if HAS_ASYNC
            for (int i = 0; i < 4; ++i)
                async_cp16(Km + (baseBS + sr) * rowStride + headOff + c0 + i * 8,
                           &Ks[r][c0 + i * 8]);
#else
            const u32x4* src = (const u32x4*)(Km + (baseBS + sr) * rowStride + headOff + c0);
            for (int i = 0; i < 4; ++i) *(u32x4*)&Ks[r][c0 + i * 8] = src[i];
#endif
        }
        // ---- stage V tile transposed (sync; needs in-register transpose) ----
        {
            const int dk0 = (tid & 7) * 8;
            const int kvq = tid >> 3;
            Pack8 rows[4];
            for (int j = 0; j < 4; ++j) {
                const int s = kv0 + kvq * 4 + j;
                const size_t sr = (size_t)(s < Sdim ? s : 0);
                rows[j].q = *(const u32x4*)(Vm + (baseBS + sr) * rowStride + headOff + dk0);
            }
            for (int i = 0; i < 8; ++i) {
                bf16x4 p;
                p.x = rows[0].e[i]; p.y = rows[1].e[i];
                p.z = rows[2].e[i]; p.w = rows[3].e[i];
                *(bf16x4*)&Vt[dk0 + i][kvq * 4] = p;
            }
        }
        WAIT_ASYNC(0);
        __syncthreads();

        // ---- scores: 16 (q) x 128 (kv) per wave ----
        v8f sacc[8];
        for (int i = 0; i < 8; ++i) sacc[i] = {};
        for (int kk = 0; kk < 2; ++kk) {
            for (int ni = 0; ni < 8; ++ni) {
                Frag kf;
                const int r   = ni * 16 + lane16;
                const int off = kk * 32 + laneHalf * 8;
                kf.q[0] = *(const u32x4*)&Ks[r][off];
                kf.q[1] = *(const u32x4*)&Ks[r][off + 16];
                sacc[ni] = __builtin_amdgcn_wmma_f32_16x16x32_bf16(
                    false, qf[kk].v, false, kf.v, (short)0, sacc[ni], false, false);
            }
        }

        // ---- online softmax per q row ----
        for (int r = 0; r < 8; ++r) {
            const int qg = q0 + wave * 16 + laneHalf * 8 + r;
            float sv[8];
            float rmax = -INFINITY;
            for (int ni = 0; ni < 8; ++ni) {
                const int kg = kv0 + ni * 16 + lane16;
                float s = sacc[ni][r] * 0.125f;   // 1/sqrt(64)
                if (kg > qg || kg >= Sdim || qg >= Sdim) s = -INFINITY;
                sv[ni] = s;
                rmax = fmaxf(rmax, s);
            }
            for (int m = 8; m >= 1; m >>= 1)
                rmax = fmaxf(rmax, __shfl_xor(rmax, m, 32));

            const float mnew = fmaxf(mrow[r], rmax);
            float scale = 1.0f, rsum = 0.0f;
            if (mnew == -INFINITY) {
                for (int ni = 0; ni < 8; ++ni)
                    Ps[wave][laneHalf * 8 + r][ni * 16 + lane16] = (bf16)0.0f;
            } else {
                scale = __expf(mrow[r] - mnew);
                for (int ni = 0; ni < 8; ++ni) {
                    const float p = __expf(sv[ni] - mnew);
                    rsum += p;
                    Ps[wave][laneHalf * 8 + r][ni * 16 + lane16] = (bf16)p;
                }
            }
            for (int m = 8; m >= 1; m >>= 1)
                rsum += __shfl_xor(rsum, m, 32);
            lrow[r] = lrow[r] * scale + rsum;
            mrow[r] = mnew;
            for (int di = 0; di < 4; ++di) oacc[di][r] *= scale;
        }

        // ---- O += P @ V ----
        for (int kk2 = 0; kk2 < 4; ++kk2) {
            Frag pf;
            {
                const int r  = lane16;
                const int ho = laneHalf * 8;
                pf.q[0] = *(const u32x4*)&Ps[wave][r][kk2 * 32 + ho];
                pf.q[1] = *(const u32x4*)&Ps[wave][r][kk2 * 32 + 16 + ho];
            }
            for (int di = 0; di < 4; ++di) {
                Frag vf;
                const int r   = di * 16 + lane16;
                const int off = kk2 * 32 + laneHalf * 8;
                vf.q[0] = *(const u32x4*)&Vt[r][off];
                vf.q[1] = *(const u32x4*)&Vt[r][off + 16];
                oacc[di] = __builtin_amdgcn_wmma_f32_16x16x32_bf16(
                    false, pf.v, false, vf.v, (short)0, oacc[di], false, false);
            }
        }
        __syncthreads();
    }

    // ---- normalize + write ----
    for (int r = 0; r < 8; ++r) {
        const int qg = q0 + wave * 16 + laneHalf * 8 + r;
        if (qg >= Sdim) continue;
        const float inv = lrow[r] > 0.0f ? 1.0f / lrow[r] : 0.0f;
        for (int di = 0; di < 4; ++di) {
            const int col = di * 16 + lane16;
            O[(baseBS + qg) * rowStride + headOff + col] = (bf16)(oacc[di][r] * inv);
        }
    }
}

// ---------------------------------------------------------------------------
// Launch
// ---------------------------------------------------------------------------
extern "C" void kernel_launch(void* const* d_in, const int* in_sizes, int n_in,
                              void* d_out, int out_size, void* d_ws, size_t ws_size,
                              hipStream_t stream)
{
    (void)in_sizes; (void)n_in; (void)out_size; (void)ws_size;

    const float* x   = (const float*)d_in[0];
    const float* wq  = (const float*)d_in[1];
    const float* wk  = (const float*)d_in[2];
    const float* wv  = (const float*)d_in[3];
    const float* wo  = (const float*)d_in[4];
    const float* bo  = (const float*)d_in[5];
    const float* w1  = (const float*)d_in[6];
    const float* b1  = (const float*)d_in[7];
    const float* w2  = (const float*)d_in[8];
    const float* b2  = (const float*)d_in[9];
    const float* g1  = (const float*)d_in[10];
    const float* be1 = (const float*)d_in[11];
    const float* g2  = (const float*)d_in[12];
    const float* be2 = (const float*)d_in[13];

    const int Dm = 1024, DFF = 4096, Bc = 8, Sc = 1000, Hh = 16;
    const int Mr = Bc * Sc;

    char* ws = (char*)d_ws;
    auto carve = [&](size_t bytes) {
        char* p = ws;
        ws += (bytes + 255) & ~(size_t)255;
        return p;
    };
    // all weights stored TRANSPOSED: [N][K]
    bf16* wq_t = (bf16*)carve((size_t)Dm * Dm  * 2);
    bf16* wk_t = (bf16*)carve((size_t)Dm * Dm  * 2);
    bf16* wv_t = (bf16*)carve((size_t)Dm * Dm  * 2);
    bf16* wo_t = (bf16*)carve((size_t)Dm * Dm  * 2);
    bf16* w1_t = (bf16*)carve((size_t)Dm * DFF * 2);
    bf16* w2_t = (bf16*)carve((size_t)Dm * DFF * 2);
    bf16* h_b  = (bf16*)carve((size_t)Mr * Dm  * 2);   // reused for h2
    bf16* q_b  = (bf16*)carve((size_t)Mr * Dm  * 2);
    bf16* k_b  = (bf16*)carve((size_t)Mr * Dm  * 2);
    bf16* v_b  = (bf16*)carve((size_t)Mr * Dm  * 2);
    bf16* a_b  = (bf16*)carve((size_t)Mr * Dm  * 2);
    float* x1  = (float*)carve((size_t)Mr * Dm * 4);
    bf16* ff_b = (bf16*)carve((size_t)Mr * DFF * 2);

    // 1) weights -> bf16, transposed to [N][K]
    dim3 tDD(Dm / 32, Dm / 32);
    cvt_transpose_bf16_kernel<<<tDD, 256, 0, stream>>>(wq, wq_t, Dm, Dm);
    cvt_transpose_bf16_kernel<<<tDD, 256, 0, stream>>>(wk, wk_t, Dm, Dm);
    cvt_transpose_bf16_kernel<<<tDD, 256, 0, stream>>>(wv, wv_t, Dm, Dm);
    cvt_transpose_bf16_kernel<<<tDD, 256, 0, stream>>>(wo, wo_t, Dm, Dm);
    dim3 tDF(DFF / 32, Dm / 32);
    cvt_transpose_bf16_kernel<<<tDF, 256, 0, stream>>>(w1, w1_t, Dm, DFF);
    dim3 tFD(Dm / 32, DFF / 32);
    cvt_transpose_bf16_kernel<<<tFD, 256, 0, stream>>>(w2, w2_t, DFF, Dm);

    // 2) LN1: x -> h (bf16)
    layernorm_bf16_kernel<<<Mr, 256, 0, stream>>>(x, g1, be1, h_b, Dm);

    // 3) Q/K/V projections (bf16 out, no bias)
    dim3 gD(Dm / 128, (Mr + 127) / 128);
    gemm_bf16_kernel<false, false, false, false, true><<<gD, 256, 0, stream>>>(
        h_b, wq_t, nullptr, nullptr, nullptr, q_b, Mr, Dm, Dm);
    gemm_bf16_kernel<false, false, false, false, true><<<gD, 256, 0, stream>>>(
        h_b, wk_t, nullptr, nullptr, nullptr, k_b, Mr, Dm, Dm);
    gemm_bf16_kernel<false, false, false, false, true><<<gD, 256, 0, stream>>>(
        h_b, wv_t, nullptr, nullptr, nullptr, v_b, Mr, Dm, Dm);

    // 4) causal attention
    dim3 gA((Sc + 127) / 128, Hh, Bc);
    attention_kernel<<<gA, 256, 0, stream>>>(q_b, k_b, v_b, a_b, Sc, Dm);

    // 5) output projection + bias + residual -> x1 (f32)
    gemm_bf16_kernel<true, false, true, true, false><<<gD, 256, 0, stream>>>(
        a_b, wo_t, bo, x, x1, nullptr, Mr, Dm, Dm);

    // 6) LN2: x1 -> h2 (bf16, reuses h_b)
    layernorm_bf16_kernel<<<Mr, 256, 0, stream>>>(x1, g2, be2, h_b, Dm);

    // 7) FFN up + bias + exact GELU -> ff (bf16)
    dim3 gF(DFF / 128, (Mr + 127) / 128);
    gemm_bf16_kernel<true, true, false, false, true><<<gF, 256, 0, stream>>>(
        h_b, w1_t, b1, nullptr, nullptr, ff_b, Mr, DFF, Dm);

    // 8) FFN down + bias + residual -> d_out (f32)
    gemm_bf16_kernel<true, false, true, true, false><<<gD, 256, 0, stream>>>(
        ff_b, w2_t, b2, x1, (float*)d_out, nullptr, Mr, Dm, DFF);
}
